// MoE_78245714198529
// MI455X (gfx1250) — compile-verified, matrix operand order checked by reading, not run on hardware
//
#include <hip/hip_runtime.h>

// ---------------- problem constants (fixed by reference) ----------------
#define BT   4096      // B*T tokens
#define DDIM 1024      // hidden D
#define EXP  8         // experts
#define FDIM 2048      // ffn F
#define NSLOT (2*BT)   // top-2 slots; slot = 2*token + k
#define APITCH 40      // LDS row pitch in bf16 elems (80B: 16B-aligned, bank-skewed)

typedef __attribute__((ext_vector_type(16))) __bf16 v16bf;
typedef __attribute__((ext_vector_type(8)))  float  v8f;

__device__ __forceinline__ unsigned short f2bf(float f) {
    unsigned u = __float_as_uint(f);
    unsigned r = (u + 0x7FFFu + ((u >> 16) & 1u)) >> 16;   // round-to-nearest-even
    return (unsigned short)r;
}

// async global->LDS copy of 16B, per-lane addresses (gfx1250 ASYNCcnt path)
#define ASYNC_LD16(ldsoff, gptr)                                                \
    asm volatile("global_load_async_to_lds_b128 %0, %1, off"                    \
                 :: "v"(ldsoff), "v"(gptr) : "memory")

// Build one 16x16x32 bf16 WMMA fragment from an LDS row (lane-local part).
// Lanes 0-15 hold K {0..7,16..23}, lanes 16-31 hold K {8..15,24..31}.
__device__ __forceinline__ v16bf lds_frag(const unsigned short* row, int koff) {
    v16bf v;
    *(uint4*)&v         = *(const uint4*)(row + koff);
    *(((uint4*)&v) + 1) = *(const uint4*)(row + koff + 16);
    return v;
}

// ---------------- prep kernels ----------------
__global__ __launch_bounds__(256) void cvt_x_kernel(const float* __restrict__ x,
                                                    unsigned short* __restrict__ xb) {
    int i = blockIdx.x * 256 + threadIdx.x;
    if (i < BT * DDIM) xb[i] = f2bf(x[i]);
}

// Transpose + convert: in [E][R][C] f32 -> out [E][C][R] bf16, 64x64 LDS tiles.
// Coalesced f32 reads, bank-padded LDS, coalesced ushort2 writes.
__global__ __launch_bounds__(256) void transpose_cvt_kernel(const float* __restrict__ in,
                                                            unsigned short* __restrict__ out,
                                                            int R, int C) {
    __shared__ unsigned short t[64][72];
    int e = blockIdx.z;
    int r0 = blockIdx.y * 64, c0 = blockIdx.x * 64;
    const float* src = in + ((size_t)e * R + r0) * C + c0;
#pragma unroll
    for (int k = 0; k < 16; ++k) {
        int idx = threadIdx.x + 256 * k;     // 0..4095
        int r = idx >> 6, c = idx & 63;
        t[c][r] = f2bf(src[(size_t)r * C + c]);
    }
    __syncthreads();
    unsigned short* dst = out + ((size_t)e * C + c0) * R + r0;
#pragma unroll
    for (int k = 0; k < 8; ++k) {
        int idx = threadIdx.x + 256 * k;     // 0..2047 element-pairs
        int c = idx >> 5, rp = (idx & 31) * 2;
        *(ushort2*)&dst[(size_t)c * R + rp] = *(const ushort2*)&t[c][rp];
    }
}

// ---------------- router: one wave per token ----------------
__global__ __launch_bounds__(128) void router_kernel(const float* __restrict__ x,
                                                     const float* __restrict__ wr,   // [D][E]
                                                     int* __restrict__ sel,          // [NSLOT]
                                                     float* __restrict__ wgt) {      // [NSLOT]
    int wave = threadIdx.x >> 5, lane = threadIdx.x & 31;
    int t = blockIdx.x * 4 + wave;
    if (t >= BT) return;
    float acc[EXP];
#pragma unroll
    for (int e = 0; e < EXP; ++e) acc[e] = 0.f;
    const float* xr = x + (size_t)t * DDIM;
    for (int d = lane; d < DDIM; d += 32) {
        float xv = xr[d];
        const float* rw = wr + (size_t)d * EXP;
#pragma unroll
        for (int e = 0; e < EXP; ++e) acc[e] += xv * rw[e];
    }
#pragma unroll
    for (int off = 16; off > 0; off >>= 1)
#pragma unroll
        for (int e = 0; e < EXP; ++e) acc[e] += __shfl_xor(acc[e], off, 32);
    if (lane == 0) {
        float m = acc[0];
#pragma unroll
        for (int e = 1; e < EXP; ++e) m = fmaxf(m, acc[e]);
        float p[EXP]; float s = 0.f;
#pragma unroll
        for (int e = 0; e < EXP; ++e) { p[e] = __expf(acc[e] - m); s += p[e]; }
        float inv = 1.f / s;
#pragma unroll
        for (int e = 0; e < EXP; ++e) p[e] *= inv;
        int i0 = 0;
#pragma unroll
        for (int e = 1; e < EXP; ++e) if (p[e] > p[i0]) i0 = e;
        int i1 = (i0 == 0) ? 1 : 0;
#pragma unroll
        for (int e = 0; e < EXP; ++e) if (e != i0 && p[e] > p[i1]) i1 = e;
        float b = __expf(p[i1] - p[i0]);
        float nw0 = 1.f / (1.f + b);
        sel[2 * t + 0] = i0;  wgt[2 * t + 0] = nw0;
        sel[2 * t + 1] = i1;  wgt[2 * t + 1] = 1.f - nw0;
    }
}

// ---------------- deterministic per-expert slot lists ----------------
__global__ __launch_bounds__(32) void build_lists_kernel(const int* __restrict__ sel,
                                                         int* __restrict__ lists,  // [E][BT]
                                                         int* __restrict__ cnt) {  // [E]
    int e = blockIdx.x;
    if (threadIdx.x != 0) return;
    int c = 0;
    for (int s = 0; s < NSLOT; ++s)
        if (sel[s] == e) lists[e * BT + (c++)] = s;
    cnt[e] = c;
}

// ---------------- grouped GEMM 1: fuse = silu(x@Wg) * (x@Wu) ----------------
// grid: (F/64, BT/128, E), block 128 (4 waves, 2x2; wave tile 64x32)
// 16 ds_load_b128 : 16 wmma per wave per k-step (fits WGP LDS bandwidth).
__global__ __launch_bounds__(128) void mlp1_kernel(const unsigned short* __restrict__ xb,    // [BT][D]
                                                   const unsigned short* __restrict__ wgT,   // [E][F][D]
                                                   const unsigned short* __restrict__ wuT,   // [E][F][D]
                                                   const int* __restrict__ lists,
                                                   const int* __restrict__ cnt,
                                                   unsigned short* __restrict__ fuse) {      // [NSLOT][F]
    int e = blockIdx.z, mt = blockIdx.y, ft = blockIdx.x;
    int ce = cnt[e];
    if (mt * 128 >= ce) return;

    __shared__ int srow[128];
    __shared__ unsigned short As[2][128][APITCH];
    __shared__ unsigned short Gs[2][64][APITCH];
    __shared__ unsigned short Us[2][64][APITCH];
    const unsigned ABUF = 128 * APITCH * 2;   // bytes per A buffer
    const unsigned BBUF = 64 * APITCH * 2;    // bytes per B buffer

    int tid = threadIdx.x;
    {
        int r = mt * 128 + tid;
        srow[tid] = (r < ce) ? lists[e * BT + r] : -1;
    }
    __syncthreads();

    // --- DMA staging: thread serves A rows r0+{0,32,64,96}, B rows r0+{0,32}, segment seg ---
    int r0  = tid >> 2;       // 0..31
    int seg = tid & 3;        // 16B chunk of a 64B row-slice
    const unsigned short* ga[4]; const unsigned short* gg[2]; const unsigned short* gu[2];
    unsigned lA0[4], lG0[2], lU0[2];
#pragma unroll
    for (int h = 0; h < 4; ++h) {
        int row = r0 + h * 32;
        int s   = srow[row];
        int tok = (s < 0) ? 0 : (s >> 1);
        ga[h]  = xb + (size_t)tok * DDIM + seg * 8;
        lA0[h] = (unsigned)(size_t)&As[0][row][seg * 8];
    }
#pragma unroll
    for (int h = 0; h < 2; ++h) {
        int row = r0 + h * 32;
        int f = ft * 64 + row;
        gg[h]  = wgT + ((size_t)e * FDIM + f) * DDIM + seg * 8;
        gu[h]  = wuT + ((size_t)e * FDIM + f) * DDIM + seg * 8;
        lG0[h] = (unsigned)(size_t)&Gs[0][row][seg * 8];
        lU0[h] = (unsigned)(size_t)&Us[0][row][seg * 8];
    }

    int wave = tid >> 5, lane = tid & 31;
    int wm = (wave >> 1) * 64, wn = (wave & 1) * 32;
    int ln = lane & 15, hi = lane >> 4;
    int koff = hi ? 8 : 0;

    v8f accg[4][2] = {}, accu[4][2] = {};

    const int NSTEP = DDIM / 32;
    // prologue: stage k-block 0 into buffer 0
#pragma unroll
    for (int h = 0; h < 4; ++h) ASYNC_LD16(lA0[h], ga[h]);
#pragma unroll
    for (int h = 0; h < 2; ++h) { ASYNC_LD16(lG0[h], gg[h]); ASYNC_LD16(lU0[h], gu[h]); }
    for (int i = 0; i < NSTEP; ++i) {
        if (i + 1 < NSTEP) {
            int kb = (i + 1) * 32;
            unsigned boffA = ((i + 1) & 1) ? ABUF : 0u;
            unsigned boffB = ((i + 1) & 1) ? BBUF : 0u;
#pragma unroll
            for (int h = 0; h < 4; ++h) ASYNC_LD16(lA0[h] + boffA, ga[h] + kb);
#pragma unroll
            for (int h = 0; h < 2; ++h) { ASYNC_LD16(lG0[h] + boffB, gg[h] + kb);
                                          ASYNC_LD16(lU0[h] + boffB, gu[h] + kb); }
            asm volatile("s_wait_asynccnt 0x8" ::: "memory");  // current stage done
        } else {
            asm volatile("s_wait_asynccnt 0x0" ::: "memory");
        }
        __syncthreads();
        int b = i & 1;
        v16bf a[4];
#pragma unroll
        for (int mi = 0; mi < 4; ++mi) a[mi] = lds_frag(&As[b][wm + mi * 16 + ln][0], koff);
        {   // gating WMMAs (one B-frag set live at a time)
            v16bf g0 = lds_frag(&Gs[b][wn + ln][0], koff);
            v16bf g1 = lds_frag(&Gs[b][wn + 16 + ln][0], koff);
#pragma unroll
            for (int mi = 0; mi < 4; ++mi) {
                accg[mi][0] = __builtin_amdgcn_wmma_f32_16x16x32_bf16(
                    false, a[mi], false, g0, (short)0, accg[mi][0], false, false);
                accg[mi][1] = __builtin_amdgcn_wmma_f32_16x16x32_bf16(
                    false, a[mi], false, g1, (short)0, accg[mi][1], false, false);
            }
        }
        {   // up WMMAs
            v16bf u0 = lds_frag(&Us[b][wn + ln][0], koff);
            v16bf u1 = lds_frag(&Us[b][wn + 16 + ln][0], koff);
#pragma unroll
            for (int mi = 0; mi < 4; ++mi) {
                accu[mi][0] = __builtin_amdgcn_wmma_f32_16x16x32_bf16(
                    false, a[mi], false, u0, (short)0, accu[mi][0], false, false);
                accu[mi][1] = __builtin_amdgcn_wmma_f32_16x16x32_bf16(
                    false, a[mi], false, u1, (short)0, accu[mi][1], false, false);
            }
        }
        __syncthreads();  // protect buffer b from next overwrite
    }

    // epilogue: silu(g)*u -> bf16 fuse[slot][f]   (v_exp + v_rcp)
#pragma unroll
    for (int mi = 0; mi < 4; ++mi)
#pragma unroll
        for (int ni = 0; ni < 2; ++ni) {
            int ncol = ft * 64 + wn + ni * 16 + ln;
#pragma unroll
            for (int r = 0; r < 8; ++r) {
                int mloc = wm + mi * 16 + r + hi * 8;
                int frow = srow[mloc];
                if (frow >= 0) {
                    float gv = accg[mi][ni][r];
                    float uv = accu[mi][ni][r];
                    float sig = __builtin_amdgcn_rcpf(1.f + __expf(-gv));
                    fuse[(size_t)frow * FDIM + ncol] = f2bf(gv * sig * uv);
                }
            }
        }
}

// ---------------- grouped GEMM 2: dslot = fuse @ Wd ----------------
// grid: (D/64, BT/128, E), block 128; wave tile 64x32; async double-buffered
__global__ __launch_bounds__(128) void mlp2_kernel(const unsigned short* __restrict__ fuse,  // [NSLOT][F]
                                                   const unsigned short* __restrict__ wdT,   // [E][D][F]
                                                   const int* __restrict__ lists,
                                                   const int* __restrict__ cnt,
                                                   float* __restrict__ dslot) {              // [NSLOT][D]
    int e = blockIdx.z, mt = blockIdx.y, dt = blockIdx.x;
    int ce = cnt[e];
    if (mt * 128 >= ce) return;

    __shared__ int srow[128];
    __shared__ unsigned short As[2][128][APITCH];
    __shared__ unsigned short Bs[2][64][APITCH];
    const unsigned ABUF = 128 * APITCH * 2;
    const unsigned BBUF = 64 * APITCH * 2;

    int tid = threadIdx.x;
    {
        int r = mt * 128 + tid;
        srow[tid] = (r < ce) ? lists[e * BT + r] : -1;
    }
    __syncthreads();

    int r0  = tid >> 2;
    int seg = tid & 3;
    const unsigned short* ga[4]; const unsigned short* gb[2];
    unsigned lA0[4], lB0[2];
#pragma unroll
    for (int h = 0; h < 4; ++h) {
        int row = r0 + h * 32;
        int s   = srow[row];
        int fr  = (s < 0) ? 0 : s;
        ga[h]  = fuse + (size_t)fr * FDIM + seg * 8;
        lA0[h] = (unsigned)(size_t)&As[0][row][seg * 8];
    }
#pragma unroll
    for (int h = 0; h < 2; ++h) {
        int row = r0 + h * 32;
        int d = dt * 64 + row;
        gb[h]  = wdT + ((size_t)e * DDIM + d) * FDIM + seg * 8;
        lB0[h] = (unsigned)(size_t)&Bs[0][row][seg * 8];
    }

    int wave = tid >> 5, lane = tid & 31;
    int wm = (wave >> 1) * 64, wn = (wave & 1) * 32;
    int ln = lane & 15, hi = lane >> 4;
    int koff = hi ? 8 : 0;

    v8f acc[4][2] = {};

    const int NSTEP = FDIM / 32;
#pragma unroll
    for (int h = 0; h < 4; ++h) ASYNC_LD16(lA0[h], ga[h]);
#pragma unroll
    for (int h = 0; h < 2; ++h) ASYNC_LD16(lB0[h], gb[h]);
    for (int i = 0; i < NSTEP; ++i) {
        if (i + 1 < NSTEP) {
            int kb = (i + 1) * 32;
            unsigned boffA = ((i + 1) & 1) ? ABUF : 0u;
            unsigned boffB = ((i + 1) & 1) ? BBUF : 0u;
#pragma unroll
            for (int h = 0; h < 4; ++h) ASYNC_LD16(lA0[h] + boffA, ga[h] + kb);
#pragma unroll
            for (int h = 0; h < 2; ++h) ASYNC_LD16(lB0[h] + boffB, gb[h] + kb);
            asm volatile("s_wait_asynccnt 0x6" ::: "memory");
        } else {
            asm volatile("s_wait_asynccnt 0x0" ::: "memory");
        }
        __syncthreads();
        int b = i & 1;
        v16bf a[4];
#pragma unroll
        for (int mi = 0; mi < 4; ++mi) a[mi] = lds_frag(&As[b][wm + mi * 16 + ln][0], koff);
        v16bf b0 = lds_frag(&Bs[b][wn + ln][0], koff);
        v16bf b1 = lds_frag(&Bs[b][wn + 16 + ln][0], koff);
#pragma unroll
        for (int mi = 0; mi < 4; ++mi) {
            acc[mi][0] = __builtin_amdgcn_wmma_f32_16x16x32_bf16(
                false, a[mi], false, b0, (short)0, acc[mi][0], false, false);
            acc[mi][1] = __builtin_amdgcn_wmma_f32_16x16x32_bf16(
                false, a[mi], false, b1, (short)0, acc[mi][1], false, false);
        }
        __syncthreads();
    }

#pragma unroll
    for (int mi = 0; mi < 4; ++mi)
#pragma unroll
        for (int ni = 0; ni < 2; ++ni) {
            int ncol = dt * 64 + wn + ni * 16 + ln;
#pragma unroll
            for (int r = 0; r < 8; ++r) {
                int mloc = wm + mi * 16 + r + hi * 8;
                int frow = srow[mloc];
                if (frow >= 0)
                    dslot[(size_t)frow * DDIM + ncol] = acc[mi][ni][r];
            }
        }
}

// ---------------- weighted combine (also initializes out) ----------------
__global__ __launch_bounds__(256) void combine_kernel(const float* __restrict__ dslot,
                                                      const float* __restrict__ wgt,
                                                      float* __restrict__ out) {
    int i = blockIdx.x * 256 + threadIdx.x;
    if (i >= BT * DDIM) return;
    int t = i >> 10;           // DDIM = 1024
    int d = i & 1023;
    float w0 = wgt[2 * t], w1 = wgt[2 * t + 1];
    out[i] = w0 * dslot[(size_t)(2 * t) * DDIM + d]
           + w1 * dslot[(size_t)(2 * t + 1) * DDIM + d];
}

// ---------------- host launcher ----------------
extern "C" void kernel_launch(void* const* d_in, const int* in_sizes, int n_in,
                              void* d_out, int out_size, void* d_ws, size_t ws_size,
                              hipStream_t stream) {
    const float* x  = (const float*)d_in[0];
    const float* wr = (const float*)d_in[1];
    const float* wg = (const float*)d_in[2];
    const float* wu = (const float*)d_in[3];
    const float* wd = (const float*)d_in[4];
    float* out = (float*)d_out;

    char* ws = (char*)d_ws;
    size_t off = 0;
    auto take = [&](size_t bytes) { char* p = ws + off; off = (off + bytes + 255) & ~(size_t)255; return p; };

    unsigned short* xb   = (unsigned short*)take((size_t)BT * DDIM * 2);
    unsigned short* wgT  = (unsigned short*)take((size_t)EXP * FDIM * DDIM * 2);
    unsigned short* wuT  = (unsigned short*)take((size_t)EXP * FDIM * DDIM * 2);
    unsigned short* wdT  = (unsigned short*)take((size_t)EXP * DDIM * FDIM * 2);
    unsigned short* fuse = (unsigned short*)take((size_t)NSLOT * FDIM * 2);
    float*          dsl  = (float*)take((size_t)NSLOT * DDIM * 4);
    int*            sel  = (int*)take((size_t)NSLOT * 4);
    float*          wgt  = (float*)take((size_t)NSLOT * 4);
    int*            lst  = (int*)take((size_t)EXP * BT * 4);
    int*            cnt  = (int*)take((size_t)EXP * 4);

    // 1) precision/layout prep
    cvt_x_kernel<<<(BT * DDIM + 255) / 256, 256, 0, stream>>>(x, xb);
    {
        dim3 gDF(FDIM / 64, DDIM / 64, EXP);   // [E][D][F] -> [E][F][D]
        transpose_cvt_kernel<<<gDF, 256, 0, stream>>>(wg, wgT, DDIM, FDIM);
        transpose_cvt_kernel<<<gDF, 256, 0, stream>>>(wu, wuT, DDIM, FDIM);
        dim3 gFD(DDIM / 64, FDIM / 64, EXP);   // [E][F][D] -> [E][D][F]
        transpose_cvt_kernel<<<gFD, 256, 0, stream>>>(wd, wdT, FDIM, DDIM);
    }
    // 2) router + 3) deterministic lists
    router_kernel<<<BT / 4, 128, 0, stream>>>(x, wr, sel, wgt);
    build_lists_kernel<<<EXP, 32, 0, stream>>>(sel, lst, cnt);
    // 4) gating/up grouped GEMM (+silu*mul fusion)
    {
        dim3 grid(FDIM / 64, BT / 128, EXP);
        mlp1_kernel<<<grid, 128, 0, stream>>>(xb, wgT, wuT, lst, cnt, fuse);
    }
    // 5) down grouped GEMM
    {
        dim3 grid(DDIM / 64, BT / 128, EXP);
        mlp2_kernel<<<grid, 128, 0, stream>>>(fuse, wdT, lst, cnt, dsl);
    }
    // 6) weighted combine -> out
    combine_kernel<<<(BT * DDIM + 255) / 256, 256, 0, stream>>>(dsl, wgt, out);
}
